// SelfAttention_51951924412772
// MI455X (gfx1250) — matrix operand order repaired and checked by reading, not compiled, not run on hardware
//
#include <hip/hip_runtime.h>

typedef __attribute__((ext_vector_type(16))) _Float16 v16h;
typedef __attribute__((ext_vector_type(8)))  float    v8f;

#define BN  4
#define NN  4096
#define CC  64
#define DKK 8
#define LOG2E 1.44269504088896340736f
#define VPITCH 40   // V row pitch in halves: 80B, 16B-aligned, low-conflict

union Frag16 { v16h h; unsigned u[8]; };

static __device__ __forceinline__ unsigned pack2(float a, float b) {
#if __has_builtin(__builtin_amdgcn_cvt_pkrtz)
    auto p = __builtin_amdgcn_cvt_pkrtz(a, b);     // v_cvt_pk_rtz_f16_f32
    return __builtin_bit_cast(unsigned, p);
#else
    union { _Float16 hh[2]; unsigned uu; } cv;
    cv.hh[0] = (_Float16)a; cv.hh[1] = (_Float16)b;
    return cv.uu;
#endif
}

// Raw v_exp_f32, no denormal-range guard. Softmax args are <= 0; anything
// below -126 represents a probability < 2^-126 where flush-to-zero is fine.
static __device__ __forceinline__ float fast_exp2(float x) {
#if __has_builtin(__builtin_amdgcn_exp2f)
    return __builtin_amdgcn_exp2f(x);
#else
    return __builtin_exp2f(x);
#endif
}

// ---------------------------------------------------------------------------
// Projection: f = x@Wf (keys), g = x@Wg (queries, pre-scaled by log2(e) so
// the attention softmax can run in the exp2 domain), v = x@Wh.
// Emits f16: Kh[B][N][8], Qh[B][N][8], Vth[B][64][N] (transposed).
// ---------------------------------------------------------------------------
__global__ __launch_bounds__(256) void proj_kernel(
    const float* __restrict__ x,
    const float* __restrict__ Wf, const float* __restrict__ bf,
    const float* __restrict__ Wg, const float* __restrict__ bg,
    const float* __restrict__ Wh, const float* __restrict__ bh,
    _Float16* __restrict__ Kh, _Float16* __restrict__ Qh,
    _Float16* __restrict__ Vth)
{
    __shared__ float xs[32][CC];
    const int pos0 = blockIdx.x * 32;
    for (int i = threadIdx.x; i < 32 * CC; i += 256)
        xs[i >> 6][i & 63] = x[pos0 * CC + i];
    __syncthreads();

    for (int oi = threadIdx.x; oi < 32 * 80; oi += 256) {
        const int p = oi / 80, j = oi % 80;
        const float* xr = xs[p];
        const int gpos = pos0 + p;                    // b*N + n
        if (j < 8) {                                  // f -> keys
            float acc = bf[j];
#pragma unroll
            for (int k = 0; k < CC; ++k) acc += xr[k] * Wf[k * DKK + j];
            Kh[gpos * DKK + j] = (_Float16)acc;
        } else if (j < 16) {                          // g -> queries (log2e-folded)
            const int jj = j - 8;
            float acc = bg[jj];
#pragma unroll
            for (int k = 0; k < CC; ++k) acc += xr[k] * Wg[k * DKK + jj];
            Qh[gpos * DKK + jj] = (_Float16)(acc * LOG2E);
        } else {                                      // v, stored transposed
            const int c = j - 16;
            float acc = bh[c];
#pragma unroll
            for (int k = 0; k < CC; ++k) acc += xr[k] * Wh[k * CC + c];
            const int b = gpos >> 12, n = gpos & (NN - 1);
            Vth[(b * CC + c) * NN + n] = (_Float16)acc;
        }
    }
}

// ---------------------------------------------------------------------------
// Flash attention, 4 waves/block, 1 wave = 16 query rows.
// Keys streamed in 32-wide blocks, cooperatively staged (K 512B + V 4KB)
// into double-buffered LDS: one __syncthreads per iteration, block i+1
// loads overlap block i compute. Softmax runs in exp2 domain.
//   S^T  = K_tile(16x8->32) x Q^T(32x16)   : 2x v_wmma_f32_16x16x32_f16
//   O^T += V^T(16x32) x P^T(32x16)         : 4x v_wmma_f32_16x16x32_f16
// ---------------------------------------------------------------------------
__global__ __launch_bounds__(128) void attn_kernel(
    const _Float16* __restrict__ Qh,
    const _Float16* __restrict__ Kh,
    const _Float16* __restrict__ Vth,
    float* __restrict__ out)
{
    __shared__ __align__(16) _Float16 Kl[2][32][DKK];      // 2 x 512B
    __shared__ __align__(16) _Float16 Vl[2][CC][VPITCH];   // 2 x 5120B

    const int t    = threadIdx.x;
    const int lane = t & 31;
    const int wv   = t >> 5;
    const int tile = blockIdx.x * 4 + wv;   // 1024 tiles = 4 batches * 256
    const int b    = blockIdx.x >> 6;       // uniform per block
    const int m0   = (tile & 255) << 4;
    const int q    = lane & 15;             // query column owned by this lane
    const int h    = lane >> 4;             // lane-half selects K sub-range

    const int vc  = t >> 1;                 // staging: channel row 0..63
    const int seg = t & 1;                  // staging: 16-key segment

    const v8f zero = {};

    // B operand: Q^T (32x16). Head dim 0..7 -> VGPR0..3, lanes 0..15 only.
    Frag16 bq = {};
    if (h == 0) {
        const uint4 tq = *(const uint4*)&Qh[(b * NN + m0 + q) * DKK];
        bq.u[0] = tq.x; bq.u[1] = tq.y; bq.u[2] = tq.z; bq.u[3] = tq.w;
    }

    v8f ot[4] = {zero, zero, zero, zero};
    float mrow = -__builtin_inff();         // running max, log2 domain
    float lrow = 0.0f;

    // ---- prologue: stage key-block 0 into buffer 0 ----
    {
        ((unsigned*)Kl[0])[t] = ((const unsigned*)&Kh[(b * NN) * DKK])[t];
        const uint4* src = (const uint4*)&Vth[(b * CC + vc) * NN + seg * 16];
        uint4* dst = (uint4*)&Vl[0][vc][seg * 16];
        dst[0] = src[0]; dst[1] = src[1];
    }

    const int NB = NN / 32;
    for (int i = 0; i < NB; ++i) {
        __syncthreads();                    // buffer i ready; buffer i^1 free
        const int s = i & 1;

        // ---- stage next key-block into the other buffer (overlaps compute)
        if (i + 1 < NB) {
            const int kbn = (i + 1) * 32;
            ((unsigned*)Kl[s ^ 1])[t] =
                ((const unsigned*)&Kh[(b * NN + kbn) * DKK])[t];
            const uint4* src = (const uint4*)&Vth[(b * CC + vc) * NN + kbn + seg * 16];
            uint4* dst = (uint4*)&Vl[s ^ 1][vc][seg * 16];
            dst[0] = src[0]; dst[1] = src[1];
        }

        // ---- A operands: two key tiles from LDS ----
        Frag16 a0 = {}, a1 = {};
        if (h == 0) {
            const uint4 t0 = *(const uint4*)&Kl[s][q][0];
            const uint4 t1 = *(const uint4*)&Kl[s][16 + q][0];
            a0.u[0] = t0.x; a0.u[1] = t0.y; a0.u[2] = t0.z; a0.u[3] = t0.w;
            a1.u[0] = t1.x; a1.u[1] = t1.y; a1.u[2] = t1.z; a1.u[3] = t1.w;
        }
        v8f s0 = __builtin_amdgcn_wmma_f32_16x16x32_f16(false, a0.h, false, bq.h,
                                                        (short)0, zero, false, false);
        v8f s1 = __builtin_amdgcn_wmma_f32_16x16x32_f16(false, a1.h, false, bq.h,
                                                        (short)0, zero, false, false);

        // ---- online softmax (log2 domain) over this lane's query column ----
        float tmax = s0[0];
#pragma unroll
        for (int r = 1; r < 8; ++r) tmax = fmaxf(tmax, s0[r]);
#pragma unroll
        for (int r = 0; r < 8; ++r) tmax = fmaxf(tmax, s1[r]);
        tmax = fmaxf(tmax, __shfl_xor(tmax, 16, 32));
        const float mnew  = fmaxf(mrow, tmax);
        const float scale = fast_exp2(mrow - mnew);   // bare v_exp_f32

        float psum = 0.f;
        unsigned pk0[4], pk1[4];
#pragma unroll
        for (int j = 0; j < 4; ++j) {
            const float e0 = fast_exp2(s0[2 * j]     - mnew);
            const float e1 = fast_exp2(s0[2 * j + 1] - mnew);
            const float f0 = fast_exp2(s1[2 * j]     - mnew);
            const float f1 = fast_exp2(s1[2 * j + 1] - mnew);
            psum += (e0 + e1) + (f0 + f1);
            pk0[j] = pack2(e0, e1);
            pk1[j] = pack2(f0, f1);
        }
        lrow = lrow * scale + (psum + __shfl_xor(psum, 16, 32));
        mrow = mnew;

        // ---- repack P^T into B-fragment: lane-half swap via shfl_xor(16) ----
        Frag16 pb;
#pragma unroll
        for (int j = 0; j < 4; ++j) {
            const unsigned o0 = __shfl_xor(pk0[j], 16, 32);
            const unsigned o1 = __shfl_xor(pk1[j], 16, 32);
            pb.u[j]     = h ? o1 : pk0[j];   // K rows  0..7  | 16..23
            pb.u[4 + j] = h ? pk1[j] : o0;   // K rows  8..15 | 24..31
        }

        // ---- O^T[ct] = V^T x P^T (+ rescaled accumulator), V from LDS ----
#pragma unroll
        for (int ct = 0; ct < 4; ++ct) {
            ot[ct] *= scale;
            Frag16 va;
            const _Float16* vr = &Vl[s][ct * 16 + q][16 * h];
            const uint4 t0 = *(const uint4*)vr;          // keys 16h  .. 16h+7
            const uint4 t1 = *(const uint4*)(vr + 8);    // keys 16h+8.. +7
            va.u[0] = t0.x; va.u[1] = t0.y; va.u[2] = t0.z; va.u[3] = t0.w;
            va.u[4] = t1.x; va.u[5] = t1.y; va.u[6] = t1.z; va.u[7] = t1.w;
            ot[ct] = __builtin_amdgcn_wmma_f32_16x16x32_f16(false, va.h, false, pb.h,
                                                            (short)0, ot[ct], false, false);
        }
    }

    // ---- normalize + write out[b][m0+q][c], c = ct*16 + 8h + r ----
    const float inv = 1.0f / lrow;
    float* orow = &out[(b * NN + m0 + q) * CC];
#pragma unroll
    for (int ct = 0; ct < 4; ++ct)
#pragma unroll
        for (int r = 0; r < 8; ++r)
            orow[ct * 16 + 8 * h + r] = ot[ct][r] * inv;
}

extern "C" void kernel_launch(void* const* d_in, const int* in_sizes, int n_in,
                              void* d_out, int out_size, void* d_ws, size_t ws_size,
                              hipStream_t stream) {
    const float* x  = (const float*)d_in[0];
    const float* Wf = (const float*)d_in[1];
    const float* bf = (const float*)d_in[2];
    const float* Wg = (const float*)d_in[3];
    const float* bg = (const float*)d_in[4];
    const float* Wh = (const float*)d_in[5];
    const float* bh = (const float*)d_in[6];
    float* out = (float*)d_out;

    char* ws = (char*)d_ws;
    const size_t qk_bytes = (size_t)BN * NN * DKK * sizeof(_Float16);  // 256 KB
    _Float16* Kh  = (_Float16*)(ws);
    _Float16* Qh  = (_Float16*)(ws + qk_bytes);
    _Float16* Vth = (_Float16*)(ws + 2 * qk_bytes);                    // 2 MB

    proj_kernel<<<(BN * NN) / 32, 256, 0, stream>>>(x, Wf, bf, Wg, bg, Wh, bh,
                                                    Kh, Qh, Vth);
    attn_kernel<<<(BN * NN / 16) / 4, 128, 0, stream>>>(Qh, Kh, Vth, out);
}
